// BSDESolver_54348516163986
// MI455X (gfx1250) — matrix operand order repaired and checked by reading, not compiled, not run on hardware
//
#include <hip/hip_runtime.h>
#include <hip/hip_bf16.h>

typedef __attribute__((ext_vector_type(16))) _Float16 v16h;
typedef __attribute__((ext_vector_type(8)))  float    v8f;

#define MSTEPS 50
#define PATHS_PER_WAVE 16
#define WAVES_PER_BLOCK 8
#define PATHS_PER_BLOCK 128
#define BLOCK_THREADS 256

#if __has_builtin(__builtin_amdgcn_tanhf)
#define HW_TANH 1
#endif

__device__ __forceinline__ float fast_tanh(float x) {
#ifdef HW_TANH
    return __builtin_amdgcn_tanhf(x);           // v_tanh_f32 (TRANS32)
#else
    // tanh(x) = 1 - 2/(e^{2x}+1); e^{2x} = 2^{x * 2*log2(e)}
    float a  = fminf(fmaxf(2.885390082f * x, -24.0f), 24.0f);
    float e2 = __builtin_amdgcn_exp2f(a);       // v_exp_f32
    return fmaf(-2.0f, __builtin_amdgcn_rcpf(e2 + 1.0f), 1.0f);  // v_rcp_f32
#endif
}

__device__ __forceinline__ float fast_sigmoid(float x) {
#ifdef HW_TANH
    return fmaf(0.5f, __builtin_amdgcn_tanhf(0.5f * x), 0.5f);
#else
    float a = fminf(fmaxf(-1.442695041f * x, -24.0f), 24.0f);
    return __builtin_amdgcn_rcpf(1.0f + __builtin_amdgcn_exp2f(a));
#endif
}

__global__ __launch_bounds__(BLOCK_THREADS) void bsde_wmma_kernel(
    const float* __restrict__ dw,
    const float* __restrict__ W1, const float* __restrict__ b1,
    const float* __restrict__ W2, const float* __restrict__ b2,
    const float* __restrict__ W3, const float* __restrict__ b3,
    const float* __restrict__ Y0p,
    float* __restrict__ out, int Btot)
{
    const float S0i = 1.0f / 100.0f;
    const float S0  = 100.0f;
    const float Rr  = 0.05f;
    const float SIG = 0.2f;
    const float DT  = 1.0f / (float)MSTEPS;

    __shared__ __align__(16) float lds_dw[PATHS_PER_BLOCK * MSTEPS];

    // Stage this block's contiguous dw slab (128 paths x 50 steps) into LDS.
    {
        const float4* src = reinterpret_cast<const float4*>(
            dw + (size_t)blockIdx.x * (PATHS_PER_BLOCK * MSTEPS));
        float4* dst = reinterpret_cast<float4*>(lds_dw);
        const int n4 = (PATHS_PER_BLOCK * MSTEPS) / 4; // 1600
        for (int i = threadIdx.x; i < n4; i += BLOCK_THREADS) dst[i] = src[i];
    }
    __syncthreads();

    const int lane = threadIdx.x & 31;
    const int wv   = threadIdx.x >> 5;
    const int half = lane >> 4;   // 0: lanes 0-15, 1: lanes 16-31
    const int l16  = lane & 15;   // path within the wave's 16-path tile

    // ---------- loop-invariant per-lane constants ----------
    // WMMA A tiles = W2^T : A[M=n][K=k] = W2[k][n]
    // A k order used here (contiguous, same as Bm): element e -> K = e + 16*half, row M = l16.
    v16h A0, A1;
    #pragma unroll
    for (int e = 0; e < 16; ++e) {
        int K = e + 16 * half;
        A0[e] = (_Float16)W2[K * 32 + l16];        // output units 0..15
        A1[e] = (_Float16)W2[K * 32 + 16 + l16];   // output units 16..31
    }
    // C operand preloaded with b2 (D/C layout: VGPR r -> unit r + 8*half).
    v8f cb0, cb1;
    float w3a[8], w3b[8];
    #pragma unroll
    for (int r = 0; r < 8; ++r) {
        int u0 = r + 8 * half;
        cb0[r] = b2[u0];
        cb1[r] = b2[16 + u0];
        w3a[r] = W3[u0];
        w3b[r] = W3[16 + u0];
    }
    // Layer-1 weights for this lane's hidden units (B layout: unit u = e + 16*half).
    float w10[16], w11[16], b1v[16];
    #pragma unroll
    for (int e = 0; e < 16; ++e) {
        int u = e + 16 * half;
        w10[e] = W1[u];        // W1[0][u]
        w11[e] = W1[32 + u];   // W1[1][u]
        b1v[e] = b1[u];
    }
    const float b3s = b3[0];

    // Path state, mirrored in lane pairs (l, l+16) so EXEC stays all-ones.
    float S = S0;
    float Y = Y0p[0];

    const float* mydw = lds_dw + (wv * PATHS_PER_WAVE + l16) * MSTEPS;

    #pragma unroll 1
    for (int m = 0; m < MSTEPS; ++m) {
        float dwi = mydw[m];                 // LDS, conflict-free + half-broadcast
        float tm  = (float)m * DT;           // == t_grid[:, m] (broadcast arange)
        float x0  = S * S0i;

        // Layer 1: h1 = tanh(x @ W1 + b1), produced directly in WMMA B layout.
        v16h Bm;
        #pragma unroll
        for (int e = 0; e < 16; ++e) {
            float a = fmaf(x0, w10[e], fmaf(tm, w11[e], b1v[e]));
            Bm[e] = (_Float16)fast_tanh(a);
        }

        // Layer 2: h2^T = W2^T @ h1^T + b2 -- two 16x16x32 f16 WMMAs.
        v8f d0 = __builtin_amdgcn_wmma_f32_16x16x32_f16(
            false, A0, false, Bm, (short)0, cb0, false, false);
        v8f d1 = __builtin_amdgcn_wmma_f32_16x16x32_f16(
            false, A1, false, Bm, (short)0, cb1, false, false);

        // Layer 3: z = sigmoid(tanh(h2) . W3 + b3); partial dot per lane,
        // then lane-pair reduce with ds_swizzle SWAPX16 (lanes l <-> l+16).
        float part = 0.0f;
        #pragma unroll
        for (int r = 0; r < 8; ++r) {
            part = fmaf(fast_tanh(d0[r]), w3a[r], part);
            part = fmaf(fast_tanh(d1[r]), w3b[r], part);
        }
        float other = __int_as_float(
            __builtin_amdgcn_ds_swizzle(__float_as_int(part), 0x401F));
        float z = fast_sigmoid(part + other + b3s);

        // Euler update (computed identically in both half-lanes).
        float dS = S * fmaf(SIG, dwi, Rr * DT);
        Y = Y + Rr * (Y - z * S) * DT + z * dS;
        S = S + dS;
    }

    if (lane < 16) {
        int p = blockIdx.x * PATHS_PER_BLOCK + wv * PATHS_PER_WAVE + l16;
        out[p]        = Y;   // Y output block
        out[Btot + p] = S;   // S output block
    }
}

extern "C" void kernel_launch(void* const* d_in, const int* in_sizes, int n_in,
                              void* d_out, int out_size, void* d_ws, size_t ws_size,
                              hipStream_t stream) {
    // Input order: dw, t_grid, W1, b1, W2, b2, W3, b3, Y0
    const float* dw  = (const float*)d_in[0];
    // d_in[1] (t_grid) is broadcast(arange(M)*DT): recomputed on the fly.
    const float* W1  = (const float*)d_in[2];
    const float* b1  = (const float*)d_in[3];
    const float* W2  = (const float*)d_in[4];
    const float* b2  = (const float*)d_in[5];
    const float* W3  = (const float*)d_in[6];
    const float* b3  = (const float*)d_in[7];
    const float* Y0  = (const float*)d_in[8];
    float* out = (float*)d_out;

    const int Btot = in_sizes[0] / MSTEPS;           // 1048576
    const int grid = Btot / PATHS_PER_BLOCK;          // 8192 blocks of 256

    bsde_wmma_kernel<<<grid, BLOCK_THREADS, 0, stream>>>(
        dw, W1, b1, W2, b2, W3, b3, Y0, out, Btot);
}
